// VoxtralCodecDecoder_24730421690873
// MI455X (gfx1250) — compile-verified, matrix operand order checked by reading, not run on hardware
//
#include <hip/hip_runtime.h>
#include <hip/hip_bf16.h>
#include <math.h>

// ---------------------------------------------------------------------------
// Voxtral codec decoder for gfx1250 (CDNA5, wave32, WMMA).
// GEMMs: v_wmma_f32_16x16x32_bf16, double-buffered LDS pipeline, fp32->bf16
// conversion in-register (weights are single-use per launch, so this is the
// minimal-HBM-traffic choice). Conv staging uses GLOBAL_LOAD_ASYNC_TO_LDS
// (ASYNCcnt) on the gfx1250 async path. Attention windows are <=16 keys ->
// VALU kernel.
// ---------------------------------------------------------------------------

typedef __bf16 bf16_t;
typedef __attribute__((ext_vector_type(16))) __bf16 v16bf;
typedef __attribute__((ext_vector_type(8)))  float  v8f;

union AFrag { v16bf v; bf16_t e[16]; };
union CAcc  { v8f   v; float  e[8];  };

#define HDIM 1024
#define NHEAD 8
#define HEADD 128
#define FFDIM 4096
#define NBATCH 2
#define T0 128

#define BM 128
#define BN 64
#define BK 32

// ---------------------------------------------------------------------------
// GEMM: C[M,N] = A[M,K] @ W[N,K]^T   (A,W fp32 row-major, bf16 WMMA, f32 acc)
// epi==0: C = acc ; epi==1: C = resid + scale[n]*acc  (layer-scale residual)
// Double-buffered LDS; global loads for slab k+1 issued before WMMAs of k.
// ---------------------------------------------------------------------------
__launch_bounds__(256)
__global__ void gemm_bf16_wmma(const float* __restrict__ A,
                               const float* __restrict__ W,
                               float* __restrict__ C,
                               int M, int N, int K,
                               int epi,
                               const float* __restrict__ scale,
                               const float* __restrict__ resid)
{
    __shared__ bf16_t As[2][BM][BK];   // 2 x 8 KB
    __shared__ bf16_t Ws[2][BN][BK];   // 2 x 4 KB

    const int tid  = threadIdx.x;
    const int lane = tid & 31;
    const int wid  = tid >> 5;
    const int wm   = (wid & 3) * 32;   // wave macro-tile row (0..96)
    const int wn   = (wid >> 2) * 32;  // wave macro-tile col (0..32)
    const int blockM = blockIdx.y * BM;
    const int blockN = blockIdx.x * BN;

    CAcc acc[2][2];
#pragma unroll
    for (int i = 0; i < 2; ++i)
#pragma unroll
        for (int j = 0; j < 2; ++j)
#pragma unroll
            for (int e = 0; e < 8; ++e) acc[i][j].e[e] = 0.0f;

    // register staging: all global loads issue back-to-back (no per-load wait)
    float4 ra[4];  // A tile: 128x32 f32 = 1024 float4 / 256 thr = 4 each
    float4 rw[2];  // W tile:  64x32 f32 =  512 float4 / 256 thr = 2 each

    auto load_regs = [&](int k0) {
#pragma unroll
        for (int it = 0; it < 4; ++it) {
            int idx = tid + it * 256, r = idx >> 3, c4 = (idx & 7) * 4;
            ra[it] = *(const float4*)(A + (size_t)(blockM + r) * K + k0 + c4);
        }
#pragma unroll
        for (int it = 0; it < 2; ++it) {
            int idx = tid + it * 256, r = idx >> 3, c4 = (idx & 7) * 4;
            rw[it] = *(const float4*)(W + (size_t)(blockN + r) * K + k0 + c4);
        }
    };
    auto stage_lds = [&](int buf) {
#pragma unroll
        for (int it = 0; it < 4; ++it) {
            int idx = tid + it * 256, r = idx >> 3, c4 = (idx & 7) * 4;
            As[buf][r][c4 + 0] = (bf16_t)ra[it].x;
            As[buf][r][c4 + 1] = (bf16_t)ra[it].y;
            As[buf][r][c4 + 2] = (bf16_t)ra[it].z;
            As[buf][r][c4 + 3] = (bf16_t)ra[it].w;
        }
#pragma unroll
        for (int it = 0; it < 2; ++it) {
            int idx = tid + it * 256, r = idx >> 3, c4 = (idx & 7) * 4;
            Ws[buf][r][c4 + 0] = (bf16_t)rw[it].x;
            Ws[buf][r][c4 + 1] = (bf16_t)rw[it].y;
            Ws[buf][r][c4 + 2] = (bf16_t)rw[it].z;
            Ws[buf][r][c4 + 3] = (bf16_t)rw[it].w;
        }
    };

    load_regs(0);

    const int rsel = lane & 15;
    const int kb   = (lane >> 4) * 8;
    int buf = 0;

    for (int k0 = 0; k0 < K; k0 += BK) {
        stage_lds(buf);
        __syncthreads();

        // kick off next slab's global loads before doing this slab's math
        if (k0 + BK < K) {
            load_regs(k0 + BK);
            if (k0 + 2 * BK < K) {
                __builtin_prefetch(A + (size_t)(blockM + (tid >> 1)) * K + k0 + 2 * BK, 0, 1);
                __builtin_prefetch(W + (size_t)(blockN + (tid >> 2)) * K + k0 + 2 * BK, 0, 1);
            }
        }

        // Fragments per ISA 16-bit A layout:
        // lanes 0-15: row=lane,    K 0..7 / 16..23
        // lanes 16-31: row=lane-16, K 8..15 / 24..31
        AFrag afrag[2], wfrag[2];
#pragma unroll
        for (int i = 0; i < 2; ++i) {
            int r = wm + i * 16 + rsel;
#pragma unroll
            for (int x = 0; x < 8; ++x) {
                afrag[i].e[x]     = As[buf][r][kb + x];
                afrag[i].e[8 + x] = As[buf][r][kb + 16 + x];
            }
        }
#pragma unroll
        for (int j = 0; j < 2; ++j) {
            int r = wn + j * 16 + rsel;
#pragma unroll
            for (int x = 0; x < 8; ++x) {
                wfrag[j].e[x]     = Ws[buf][r][kb + x];
                wfrag[j].e[8 + x] = Ws[buf][r][kb + 16 + x];
            }
        }
#pragma unroll
        for (int i = 0; i < 2; ++i)
#pragma unroll
            for (int j = 0; j < 2; ++j)
                acc[i][j].v = __builtin_amdgcn_wmma_f32_16x16x32_bf16(
                    false, afrag[i].v, false, wfrag[j].v,
                    (short)0, acc[i][j].v, false, false);

        buf ^= 1;
    }

    // C/D layout: element e -> row m0+e (lanes 0-15) / m0+8+e (lanes 16-31),
    // col n0 + (lane&15).
    const int rhi = (lane >> 4) * 8;
#pragma unroll
    for (int i = 0; i < 2; ++i) {
#pragma unroll
        for (int j = 0; j < 2; ++j) {
            int col = blockN + wn + j * 16 + rsel;
#pragma unroll
            for (int e = 0; e < 8; ++e) {
                int row  = blockM + wm + i * 16 + rhi + e;
                size_t o = (size_t)row * N + col;
                float r  = acc[i][j].e[e];
                if (epi == 1) r = resid[o] + scale[col] * r;
                C[o] = r;
            }
        }
    }
}

// ---------------------------------------------------------------------------
// RMSNorm over D=1024 per row; safe in-place.
// ---------------------------------------------------------------------------
__launch_bounds__(256)
__global__ void rmsnorm_kernel(const float* __restrict__ x,
                               const float* __restrict__ w,
                               float* __restrict__ y, int D)
{
    __shared__ float red[256];
    const int row = blockIdx.x;
    const float* xr = x + (size_t)row * D;
    float s = 0.0f;
    for (int i = threadIdx.x; i < D; i += 256) { float v = xr[i]; s += v * v; }
    red[threadIdx.x] = s;
    __syncthreads();
    for (int st = 128; st > 0; st >>= 1) {
        if (threadIdx.x < st) red[threadIdx.x] += red[threadIdx.x + st];
        __syncthreads();
    }
    const float inv = rsqrtf(red[0] / (float)D + 0.01f);
    float* yr = y + (size_t)row * D;
    for (int i = threadIdx.x; i < D; i += 256) yr[i] = xr[i] * inv * w[i];
}

// ---------------------------------------------------------------------------
// Sliding-window causal attention with ALiBi. window <= 16 keys back.
// One block per (t, head, batch); 128 threads = head dim.
// ---------------------------------------------------------------------------
__launch_bounds__(128)
__global__ void attn_kernel(const float* __restrict__ q,
                            const float* __restrict__ k,
                            const float* __restrict__ v,
                            float* __restrict__ o,
                            int L, int window)
{
    __shared__ float red[128];
    __shared__ float pr[32];
    const int t = blockIdx.x, h = blockIdx.y, b = blockIdx.z;
    const int d = threadIdx.x;
    const size_t base = (size_t)b * L * HDIM;
    const float qd = q[base + (size_t)t * HDIM + h * HEADD + d];
    const int noff = ((window < t) ? window : t) + 1;
    const float slope = exp2f(-(float)(h + 1));   // NH=8 -> 2^-(h+1)
    const float sc = 0.08838834764831845f;        // 1/sqrt(128)

    for (int off = 0; off < noff; ++off) {
        red[d] = qd * k[base + (size_t)(t - off) * HDIM + h * HEADD + d];
        __syncthreads();
        for (int st = 64; st > 0; st >>= 1) {
            if (d < st) red[d] += red[d + st];
            __syncthreads();
        }
        if (d == 0) pr[off] = red[0] * sc - slope * (float)off;
        __syncthreads();
    }
    if (d == 0) {
        float m = -1e30f;
        for (int i = 0; i < noff; ++i) m = fmaxf(m, pr[i]);
        float s = 0.0f;
        for (int i = 0; i < noff; ++i) { float e = __expf(pr[i] - m); pr[i] = e; s += e; }
        float inv = 1.0f / s;
        for (int i = 0; i < noff; ++i) pr[i] *= inv;
    }
    __syncthreads();
    float accv = 0.0f;
    for (int off = 0; off < noff; ++off)
        accv += pr[off] * v[base + (size_t)(t - off) * HDIM + h * HEADD + d];
    o[base + (size_t)t * HDIM + h * HEADD + d] = accv;
}

// a = silu(a) * b
__global__ void silu_mul_kernel(float* __restrict__ a, const float* __restrict__ b, size_t n)
{
    size_t i = (size_t)blockIdx.x * blockDim.x + threadIdx.x;
    if (i < n) { float x = a[i]; a[i] = (x / (1.0f + __expf(-x))) * b[i]; }
}

// torch weight_norm dim=0: w[o,:] = g[o] * v[o,:] * rsqrt(sum(v[o,:]^2))
__launch_bounds__(256)
__global__ void weight_norm_kernel(const float* __restrict__ v,
                                   const float* __restrict__ g,
                                   float* __restrict__ w, int per)
{
    __shared__ float red[256];
    const int o = blockIdx.x;
    const float* vr = v + (size_t)o * per;
    float s = 0.0f;
    for (int i = threadIdx.x; i < per; i += 256) { float x = vr[i]; s += x * x; }
    red[threadIdx.x] = s;
    __syncthreads();
    for (int st = 128; st > 0; st >>= 1) {
        if (threadIdx.x < st) red[threadIdx.x] += red[threadIdx.x + st];
        __syncthreads();
    }
    const float scl = g[o] * rsqrtf(red[0]);
    float* wr = w + (size_t)o * per;
    for (int i = threadIdx.x; i < per; i += 256) wr[i] = vr[i] * scl;
}

// codes -> x0 [B,292,T] channel-major
__launch_bounds__(320)
__global__ void dequant_kernel(const int* __restrict__ codes,
                               const float* __restrict__ emb_sum,
                               const float* __restrict__ usage,
                               float* __restrict__ x0, int T)
{
    const int bt = blockIdx.x;
    const int b = bt / T, t = bt % T;
    const int c = threadIdx.x;
    const int* cr = codes + (size_t)bt * 37;
    if (c < 256) {
        int s = cr[0] - 4;
        s = (s < 0) ? 0 : ((s > 8191) ? 8191 : s);
        float u = usage[s];
        u = fmaxf(u, 1e-8f);
        x0[((size_t)b * 292 + c) * T + t] = emb_sum[(size_t)s * 256 + c] / u;
    } else if (c < 292) {
        int ci = c - 256 + 1;
        float a = ((float)(cr[ci] - 4)) / 2047.0f * 2.0f - 1.0f;
        x0[((size_t)b * 292 + c) * T + t] = a;
    }
}

// causal conv: x[B,292,T] -> y[B,1024,T], k=3, left pad 2
__launch_bounds__(128)
__global__ void conv_in_kernel(const float* __restrict__ x,
                               const float* __restrict__ w,
                               float* __restrict__ y, int T)
{
    const int o = blockIdx.x;
    const int b = blockIdx.y;
    const int t = threadIdx.x;
    float acc = 0.0f;
    const float* wr = w + (size_t)o * 292 * 3;
    const float* xb = x + (size_t)b * 292 * T;
    for (int ci = 0; ci < 292; ++ci) {
        const float* xc = xb + (size_t)ci * T;
        const float* wc = wr + ci * 3;
#pragma unroll
        for (int kk = 0; kk < 3; ++kk) {
            int ti = t + kk - 2;
            if (ti >= 0) acc += wc[kk] * xc[ti];
        }
    }
    y[((size_t)b * 1024 + o) * T + t] = acc;
}

// [B,C,T] channel-major -> [(b*T+t), C] token-major
__global__ void chan_to_tok(const float* __restrict__ x, float* __restrict__ y,
                            int Bn, int C, int T)
{
    size_t idx = (size_t)blockIdx.x * 256 + threadIdx.x;
    size_t total = (size_t)Bn * C * T;
    if (idx >= total) return;
    int c = (int)(idx % C);
    size_t r = idx / C;
    int t = (int)(r % T);
    int b = (int)(r / T);
    y[idx] = x[((size_t)b * C + c) * T + t];
}

__global__ void tok_to_chan(const float* __restrict__ x, float* __restrict__ y,
                            int Bn, int C, int T)
{
    size_t idx = (size_t)blockIdx.x * 256 + threadIdx.x;
    size_t total = (size_t)Bn * C * T;
    if (idx >= total) return;
    int c = (int)(idx % C);
    size_t r = idx / C;
    int t = (int)(r % T);
    int b = (int)(r / T);
    y[((size_t)b * C + c) * T + t] = x[idx];
}

// causal ConvTranspose1d: k=4, stride 2, trim 2 -> out len 2T. w [I=1024,O=1024,4]
// Input columns staged into LDS with GLOBAL_LOAD_ASYNC_TO_LDS_B32 (ASYNCcnt).
// xs is the only __shared__ object -> LDS offsets are 0..4092 for col0 and
// 4096.. for col1; the VDST VGPR gives the LDS byte address.
__launch_bounds__(256)
__global__ void upsample_kernel(const float* __restrict__ x,
                                const float* __restrict__ w,
                                float* __restrict__ y, int T)
{
    __shared__ float xs[2048];
    const int tOut = blockIdx.x;
    const int b = blockIdx.y;
    const int k0 = tOut & 1;              // taps k0 and k0+2
    const int ti0 = (tOut - k0) >> 1;     // always in [0, T)
    const int ti1 = (tOut - k0 - 2) >> 1; // < 0 only for tOut in {0,1} (block-uniform)

    for (int ci = threadIdx.x; ci < 1024; ci += 256) {
        const float* g0 = x + ((size_t)b * 1024 + ci) * T + ti0;
        unsigned l0 = (unsigned)(ci * 4);
        asm volatile("global_load_async_to_lds_b32 %0, %1, off"
                     :: "v"(l0), "v"(g0) : "memory");
        if (ti1 >= 0) {
            const float* g1 = x + ((size_t)b * 1024 + ci) * T + ti1;
            unsigned l1 = (unsigned)(4096 + ci * 4);
            asm volatile("global_load_async_to_lds_b32 %0, %1, off"
                         :: "v"(l1), "v"(g1) : "memory");
        } else {
            xs[1024 + ci] = 0.0f;
        }
    }
    asm volatile("s_wait_asynccnt 0x0" ::: "memory");
    __syncthreads();

    for (int o = threadIdx.x; o < 1024; o += 256) {
        float acc = 0.0f;
        for (int ci = 0; ci < 1024; ++ci) {
            const float* wr = w + ((size_t)ci * 1024 + o) * 4;
            acc += xs[ci] * wr[k0] + xs[1024 + ci] * wr[k0 + 2];
        }
        y[((size_t)b * 1024 + o) * (2 * T) + tOut] = acc;
    }
}

// out conv k=7 causal + transpose store: out[b, t*240 + n]
// Async LDS staging when all 7 taps are in range (block-uniform condition).
__launch_bounds__(256)
__global__ void conv_out_kernel(const float* __restrict__ x,
                                const float* __restrict__ w,
                                float* __restrict__ out, int T)
{
    __shared__ float xs[7 * 1024];   // 28 KB, sole shared object -> offset 0
    const int t = blockIdx.x;
    const int b = blockIdx.y;

    if (t >= 6) {
        for (int i = threadIdx.x; i < 7 * 1024; i += 256) {
            int kk = i / 1024;
            int ci = i % 1024;
            const float* g = x + ((size_t)b * 1024 + ci) * T + (t - 6 + kk);
            unsigned l = (unsigned)(i * 4);
            asm volatile("global_load_async_to_lds_b32 %0, %1, off"
                         :: "v"(l), "v"(g) : "memory");
        }
        asm volatile("s_wait_asynccnt 0x0" ::: "memory");
    } else {
        for (int i = threadIdx.x; i < 7 * 1024; i += 256) {
            int kk = i / 1024;
            int ci = i % 1024;
            int ti = t - 6 + kk;
            xs[i] = (ti >= 0) ? x[((size_t)b * 1024 + ci) * T + ti] : 0.0f;
        }
    }
    __syncthreads();

    const int n = threadIdx.x;
    if (n < 240) {
        float acc = 0.0f;
        const float* wr = w + (size_t)n * 1024 * 7;
        for (int ci = 0; ci < 1024; ++ci) {
#pragma unroll
            for (int kk = 0; kk < 7; ++kk)
                acc += wr[ci * 7 + kk] * xs[kk * 1024 + ci];
        }
        out[(size_t)b * ((size_t)T * 240) + (size_t)t * 240 + n] = acc;
    }
}

// ---------------------------------------------------------------------------
// Host orchestration
// ---------------------------------------------------------------------------
static inline void launch_gemm(const float* A, const float* W, float* C,
                               int M, int N, int K, int epi,
                               const float* scale, const float* resid,
                               hipStream_t stream)
{
    dim3 grid(N / BN, M / BM);
    gemm_bf16_wmma<<<grid, 256, 0, stream>>>(A, W, C, M, N, K, epi, scale, resid);
}

extern "C" void kernel_launch(void* const* d_in, const int* in_sizes, int n_in,
                              void* d_out, int out_size, void* d_ws, size_t ws_size,
                              hipStream_t stream)
{
    (void)in_sizes; (void)n_in; (void)out_size; (void)ws_size;

    // ---- input mapping (setup_inputs dict order, flattened) ----
    const int*   codes     = (const int*)d_in[0];
    const float* inconv_v  = (const float*)d_in[1];
    const float* inconv_g  = (const float*)d_in[2];
    // 8 blocks x 13 tensors: wq wk wv wo q_norm k_norm attn_norm ffn_norm
    //                        attn_scale ffn_scale w1 w2 w3
    auto blk = [&](int stage, int j, int p) -> const float* {
        return (const float*)d_in[3 + (stage * 2 + j) * 13 + p];
    };
    const float* ups_v[3], * ups_g[3];
    for (int r = 0; r < 3; ++r) {
        ups_v[r] = (const float*)d_in[107 + 2 * r];
        ups_g[r] = (const float*)d_in[108 + 2 * r];
    }
    const float* outconv_v = (const float*)d_in[113];
    const float* outconv_g = (const float*)d_in[114];
    const float* emb_sum   = (const float*)d_in[115];
    const float* usage     = (const float*)d_in[116];

    // ---- workspace carve ----
    char* p = (char*)d_ws;
    auto alloc = [&](size_t bytes) -> float* {
        float* r = (float*)p;
        p += (bytes + 255) & ~(size_t)255;
        return r;
    };
    const size_t MMAX = (size_t)NBATCH * 1024;           // 2048 tokens max
    float* wn_in  = alloc((size_t)1024 * 292 * 3 * 4);
    float* wn_up0 = alloc((size_t)1024 * 1024 * 4 * 4);
    float* wn_up1 = alloc((size_t)1024 * 1024 * 4 * 4);
    float* wn_up2 = alloc((size_t)1024 * 1024 * 4 * 4);
    float* wn_out = alloc((size_t)240 * 1024 * 7 * 4);
    float* x0     = alloc((size_t)NBATCH * 292 * T0 * 4);
    float* xcA    = alloc(MMAX * HDIM * 4);              // channel-major buf
    float* xcB    = alloc(MMAX * HDIM * 4);
    float* xt     = alloc(MMAX * HDIM * 4);              // token-major residual
    float* hbuf   = alloc(MMAX * HDIM * 4);
    float* qb     = alloc(MMAX * HDIM * 4);
    float* kb     = alloc(MMAX * HDIM * 4);
    float* vb     = alloc(MMAX * HDIM * 4);
    float* ab     = alloc(MMAX * HDIM * 4);
    float* ff1    = alloc(MMAX * FFDIM * 4);
    float* ff3    = alloc(MMAX * FFDIM * 4);

    // ---- weight-norm precompute ----
    weight_norm_kernel<<<1024, 256, 0, stream>>>(inconv_v, inconv_g, wn_in, 292 * 3);
    weight_norm_kernel<<<1024, 256, 0, stream>>>(ups_v[0], ups_g[0], wn_up0, 1024 * 4);
    weight_norm_kernel<<<1024, 256, 0, stream>>>(ups_v[1], ups_g[1], wn_up1, 1024 * 4);
    weight_norm_kernel<<<1024, 256, 0, stream>>>(ups_v[2], ups_g[2], wn_up2, 1024 * 4);
    weight_norm_kernel<<<240, 256, 0, stream>>>(outconv_v, outconv_g, wn_out, 1024 * 7);
    float* wn_up[3] = { wn_up0, wn_up1, wn_up2 };

    // ---- front end ----
    dequant_kernel<<<NBATCH * T0, 320, 0, stream>>>(codes, emb_sum, usage, x0, T0);
    conv_in_kernel<<<dim3(1024, NBATCH), 128, 0, stream>>>(x0, wn_in, xcA, T0);

    static const int WINDOWS[4] = { 2, 4, 8, 16 };
    float* xc  = xcA;
    float* xc2 = xcB;

    for (int s = 0; s < 4; ++s) {
        const int T = T0 << s;
        const int M = NBATCH * T;
        const size_t elems = (size_t)M * HDIM;
        const int egrid = (int)((elems + 255) / 256);

        chan_to_tok<<<egrid, 256, 0, stream>>>(xc, xt, NBATCH, HDIM, T);

        for (int j = 0; j < 2; ++j) {
            const float* wq = blk(s, j, 0),  * wk = blk(s, j, 1);
            const float* wv = blk(s, j, 2),  * wo = blk(s, j, 3);
            const float* qn = blk(s, j, 4),  * kn = blk(s, j, 5);
            const float* an = blk(s, j, 6),  * fn = blk(s, j, 7);
            const float* asc = blk(s, j, 8), * fsc = blk(s, j, 9);
            const float* w1 = blk(s, j, 10), * w2 = blk(s, j, 11), * w3 = blk(s, j, 12);

            // attention
            rmsnorm_kernel<<<M, 256, 0, stream>>>(xt, an, hbuf, HDIM);
            launch_gemm(hbuf, wq, qb, M, HDIM, HDIM, 0, nullptr, nullptr, stream);
            launch_gemm(hbuf, wk, kb, M, HDIM, HDIM, 0, nullptr, nullptr, stream);
            launch_gemm(hbuf, wv, vb, M, HDIM, HDIM, 0, nullptr, nullptr, stream);
            rmsnorm_kernel<<<M, 256, 0, stream>>>(qb, qn, qb, HDIM);
            rmsnorm_kernel<<<M, 256, 0, stream>>>(kb, kn, kb, HDIM);
            attn_kernel<<<dim3(T, NHEAD, NBATCH), 128, 0, stream>>>(qb, kb, vb, ab,
                                                                    T, WINDOWS[s]);
            launch_gemm(ab, wo, xt, M, HDIM, HDIM, 1, asc, xt, stream);

            // SwiGLU FFN
            rmsnorm_kernel<<<M, 256, 0, stream>>>(xt, fn, hbuf, HDIM);
            launch_gemm(hbuf, w1, ff1, M, FFDIM, HDIM, 0, nullptr, nullptr, stream);
            launch_gemm(hbuf, w3, ff3, M, FFDIM, HDIM, 0, nullptr, nullptr, stream);
            {
                size_t n = (size_t)M * FFDIM;
                silu_mul_kernel<<<(int)((n + 255) / 256), 256, 0, stream>>>(ff1, ff3, n);
            }
            launch_gemm(ff1, w2, xt, M, HDIM, FFDIM, 1, fsc, xt, stream);
        }

        tok_to_chan<<<egrid, 256, 0, stream>>>(xt, xc, NBATCH, HDIM, T);

        if (s < 3) {
            upsample_kernel<<<dim3(2 * T, NBATCH), 256, 0, stream>>>(xc, wn_up[s], xc2, T);
            float* tmp = xc; xc = xc2; xc2 = tmp;
        }
    }

    // out conv at T=1024, stores transposed directly into d_out [B, T*240]
    conv_out_kernel<<<dim3(1024, NBATCH), 256, 0, stream>>>(xc, wn_out,
                                                            (float*)d_out, 1024);
}
